// BFMFB_71296457113746
// MI455X (gfx1250) — matrix operand order repaired and checked by reading, not compiled, not run on hardware
//
#include <hip/hip_runtime.h>
#include <math.h>

// ---------------------------------------------------------------------------
// MI455X (gfx1250) implementation of the frequency-domain dual-Mamba block.
// Dense algebra (FFT-as-DFT, 1x1 convs, Mamba projections, Mamba2 SSD chunk)
// -> v_wmma_f32_16x16x32_f16.  GEMM tiles double-buffer-staged with
// global_load_async_to_lds_b128 (ASYNCcnt) so DMA overlaps the WMMA stream.
// ---------------------------------------------------------------------------

typedef __attribute__((ext_vector_type(16))) _Float16 v16h;
typedef __attribute__((ext_vector_type(8)))  float    v8f;

#define EPS 1e-5f
#define ZROW 1216   // padded in_proj output row stride (1160 -> 19*64)

// ------------------------------- math helpers ------------------------------
__device__ __forceinline__ float gelu_f(float x) {
  return 0.5f * x * (1.0f + erff(x * 0.70710678118654752f));
}
__device__ __forceinline__ float silu_f(float x) { return x / (1.0f + expf(-x)); }
__device__ __forceinline__ float sigm_f(float x) { return 1.0f / (1.0f + expf(-x)); }
__device__ __forceinline__ float softplus_f(float x) {
  return (x > 20.0f) ? x : log1pf(expf(x));
}
__device__ __forceinline__ float wave_sum32(float v) {
#pragma unroll
  for (int m = 16; m >= 1; m >>= 1) v += __shfl_xor(v, m, 32);
  return v;
}

// ------------------------------ WMMA plumbing ------------------------------
__device__ __forceinline__ v8f wmma16(v16h a, v16h b, v8f c) {
  return __builtin_amdgcn_wmma_f32_16x16x32_f16(false, a, false, b, (short)0, c,
                                                false, false);
}

// f16-LDS fragment loaders (SSD kernel).
// A-operand (16x32): lane m=lane&15, half=lane>>4; VGPR q: K=(q>>2)*16+half*8+(q&3)*2
__device__ __forceinline__ v16h frag_a(const _Float16* base, int pitch, int mbase,
                                       int k0, int lane) {
  const unsigned* row =
      (const unsigned*)(base + (size_t)(mbase + (lane & 15)) * pitch + k0);
  const int hh = lane >> 4;
  union { v16h v; unsigned u[8]; } r;
#pragma unroll
  for (int q = 0; q < 8; ++q)
    r.u[q] = row[((q >> 2) << 3) + (hh << 2) + (q & 3)];
  return r.v;
}
// B-operand (32x16): lane n=lane&15, half=lane>>4; VGPR q: K=(q>>2)*16+half*8+(q&3)*2
__device__ __forceinline__ v16h frag_b(const _Float16* base, int pitch, int nbase,
                                       int k0, int lane) {
  const unsigned* row =
      (const unsigned*)(base + (size_t)(nbase + (lane & 15)) * pitch + k0);
  const int hh = lane >> 4;
  union { v16h v; unsigned u[8]; } r;
#pragma unroll
  for (int q = 0; q < 8; ++q)
    r.u[q] = row[((q >> 2) << 3) + (hh << 2) + (q & 3)];
  return r.v;
}

// f32-LDS fragment builders (async-staged GEMM): convert to f16 at read time.
__device__ __forceinline__ v16h frag_a32(const float* base, int pitch, int mbase,
                                         int k0, int lane) {
  const float* row = base + (size_t)(mbase + (lane & 15)) * pitch + k0;
  const int hh = lane >> 4;
  union { v16h v; _Float16 h[16]; } r;
#pragma unroll
  for (int q = 0; q < 8; ++q) {
    const int kb = ((q >> 2) << 4) + (hh << 3) + ((q & 3) << 1);
    r.h[2 * q]     = (_Float16)row[kb];
    r.h[2 * q + 1] = (_Float16)row[kb + 1];
  }
  return r.v;
}
__device__ __forceinline__ v16h frag_b32(const float* base, int pitch, int nbase,
                                         int k0, int lane) {
  const float* row = base + (size_t)(nbase + (lane & 15)) * pitch + k0;
  const int hh = lane >> 4;
  union { v16h v; _Float16 h[16]; } r;
#pragma unroll
  for (int q = 0; q < 8; ++q) {
    const int kb = ((q >> 2) << 4) + (hh << 3) + ((q & 3) << 1);
    r.h[2 * q]     = (_Float16)row[kb];
    r.h[2 * q + 1] = (_Float16)row[kb + 1];
  }
  return r.v;
}

// --------------------------- generic NT GEMM (f16 WMMA) --------------------
// C[M,N] = op( sum_k A[m,k] * B[n,k] ).  Caller guarantees M%64==0, N%64==0
// (B padded), K%64==0, 16B-aligned rows.  flags: bit0=accumulate, bit1=negate.
// 64x64 f32 tiles are async-DMA'd to LDS (double buffered, ASYNCcnt).
#define GPITCH 68  // f32 LDS row pitch: 272B = 17 * 16B (keeps b128 alignment)
__global__ __launch_bounds__(128) void k_gemm(const float* __restrict__ A, int lda,
                                              const float* __restrict__ B, int ldb,
                                              float* __restrict__ C, int ldc,
                                              int K, int flags) {
  __shared__ alignas(16) float Af[2][64 * GPITCH];
  __shared__ alignas(16) float Bf[2][64 * GPITCH];
  const int m0 = blockIdx.y << 6, n0 = blockIdx.x << 6;
  const int tid = threadIdx.x, lane = tid & 31, wv = tid >> 5;
  const int wm = (wv >> 1) << 5, wn = (wv & 1) << 5;
  const int colf = (tid & 15) << 2;  // 4-float segment within a 64-wide row
  const int row0 = tid >> 4;         // 0..7
  const unsigned long long baseA = (unsigned long long)(size_t)A;
  const unsigned long long baseB = (unsigned long long)(size_t)B;

  auto stage = [&](int bufi, int k0) {
#pragma unroll
    for (int j = 0; j < 8; ++j) {
      const int r = (j << 3) + row0;
      const unsigned ldsA = (unsigned)(size_t)&Af[bufi][r * GPITCH + colf];
      const unsigned offA = (unsigned)(((size_t)(m0 + r) * lda + k0 + colf) << 2);
      asm volatile("global_load_async_to_lds_b128 %0, %1, %2"
                   :: "v"(ldsA), "v"(offA), "s"(baseA) : "memory");
      const unsigned ldsB = (unsigned)(size_t)&Bf[bufi][r * GPITCH + colf];
      const unsigned offB = (unsigned)(((size_t)(n0 + r) * ldb + k0 + colf) << 2);
      asm volatile("global_load_async_to_lds_b128 %0, %1, %2"
                   :: "v"(ldsB), "v"(offB), "s"(baseB) : "memory");
    }
  };

  const v8f VZ = {0.f, 0.f, 0.f, 0.f, 0.f, 0.f, 0.f, 0.f};
  v8f acc[2][2] = {{VZ, VZ}, {VZ, VZ}};

  stage(0, 0);
  asm volatile("s_wait_asynccnt 0x0" ::: "memory");
  __syncthreads();
  for (int k0 = 0; k0 < K; k0 += 64) {
    const int cur = (k0 >> 6) & 1;
    if (k0 + 64 < K) stage(1 - cur, k0 + 64);   // overlap DMA with WMMA below
    const float* Ac = Af[cur];
    const float* Bc = Bf[cur];
#pragma unroll
    for (int ks = 0; ks < 64; ks += 32) {
      v16h a0 = frag_a32(Ac, GPITCH, wm, ks, lane);
      v16h a1 = frag_a32(Ac, GPITCH, wm + 16, ks, lane);
      v16h b0 = frag_b32(Bc, GPITCH, wn, ks, lane);
      v16h b1 = frag_b32(Bc, GPITCH, wn + 16, ks, lane);
      acc[0][0] = wmma16(a0, b0, acc[0][0]);
      acc[0][1] = wmma16(a0, b1, acc[0][1]);
      acc[1][0] = wmma16(a1, b0, acc[1][0]);
      acc[1][1] = wmma16(a1, b1, acc[1][1]);
    }
    asm volatile("s_wait_asynccnt 0x0" ::: "memory");
    __syncthreads();
  }
  // C/D layout: lane = (m>=8?16:0)+n, vgpr r -> m = half*8 + r
  const int hh = lane >> 4, nn = lane & 15;
#pragma unroll
  for (int i = 0; i < 2; ++i)
#pragma unroll
    for (int j = 0; j < 2; ++j)
#pragma unroll
      for (int r = 0; r < 8; ++r) {
        const int gm = m0 + wm + (i << 4) + (hh << 3) + r;
        const int gn = n0 + wn + (j << 4) + nn;
        float v = acc[i][j][r];
        if (flags & 2) v = -v;
        const size_t o = (size_t)gm * ldc + gn;
        if (flags & 1) C[o] += v; else C[o] = v;
      }
}

// ----------------------- Mamba2 SSD chunk kernel (WMMA) --------------------
// One block per sequence; L=64 = single chunk.  G = C*B^T is head-independent
// (B/C states are shared by all 8 heads), so it is computed once per sequence:
//   G = C * B^T                      (WMMA, once)
//   per head h: W[t,s] = (s<=t) exp(cum_h[t]-cum_h[s]) * dt_h[s] * G[t,s]
//               Y_h = W * xh_h + D_h * xh_h   (WMMA)
#define SP 72
#define GSP 65
__global__ __launch_bounds__(128) void k_ssd(const float* __restrict__ xbc,
                                             const float* __restrict__ dtb,
                                             const float* __restrict__ A_log,
                                             const float* __restrict__ Dp,
                                             float* __restrict__ yh) {
  const int n = blockIdx.x;
  __shared__ alignas(16) _Float16 Cs[64 * SP];
  __shared__ alignas(16) _Float16 Bs[64 * SP];
  __shared__ alignas(16) _Float16 Ws[64 * SP];
  __shared__ alignas(16) _Float16 xT[64 * SP];
  __shared__ alignas(16) float Gs[64 * GSP];
  __shared__ float cum[64];
  __shared__ float dts[64];
  const int tid = threadIdx.x, lane = tid & 31, wv = tid >> 5;
  const size_t base = (size_t)n * 64;

  // B/C state tiles (contiguous 16B-aligned rows), shared across heads
  for (int seg = tid; seg < 1024; seg += 128) {
    const int l = seg >> 4, c4 = (seg & 15) << 2;
    const size_t t = (base + l) * 640;
    const float4 cv = *(const float4*)(xbc + t + 576 + c4);
    const float4 bv = *(const float4*)(xbc + t + 512 + c4);
    union { unsigned long long u; _Float16 hh[4]; } pc, pb;
    pc.hh[0] = (_Float16)cv.x; pc.hh[1] = (_Float16)cv.y;
    pc.hh[2] = (_Float16)cv.z; pc.hh[3] = (_Float16)cv.w;
    pb.hh[0] = (_Float16)bv.x; pb.hh[1] = (_Float16)bv.y;
    pb.hh[2] = (_Float16)bv.z; pb.hh[3] = (_Float16)bv.w;
    *(unsigned long long*)(Cs + l * SP + c4) = pc.u;
    *(unsigned long long*)(Bs + l * SP + c4) = pb.u;
  }
  __syncthreads();

  const int wm = (wv >> 1) << 5, wn = (wv & 1) << 5;
  const int hh = lane >> 4, nn = lane & 15;
  const v8f VZ = {0.f, 0.f, 0.f, 0.f, 0.f, 0.f, 0.f, 0.f};

  // ---- GEMM1 (once): G = C * B^T ----
  {
    v8f g[2][2] = {{VZ, VZ}, {VZ, VZ}};
#pragma unroll
    for (int k0 = 0; k0 < 64; k0 += 32) {
      v16h a0 = frag_a(Cs, SP, wm, k0, lane);
      v16h a1 = frag_a(Cs, SP, wm + 16, k0, lane);
      v16h b0 = frag_b(Bs, SP, wn, k0, lane);
      v16h b1 = frag_b(Bs, SP, wn + 16, k0, lane);
      g[0][0] = wmma16(a0, b0, g[0][0]);
      g[0][1] = wmma16(a0, b1, g[0][1]);
      g[1][0] = wmma16(a1, b0, g[1][0]);
      g[1][1] = wmma16(a1, b1, g[1][1]);
    }
#pragma unroll
    for (int i = 0; i < 2; ++i)
#pragma unroll
      for (int j = 0; j < 2; ++j)
#pragma unroll
        for (int r = 0; r < 8; ++r) {
          const int t_ = wm + (i << 4) + (hh << 3) + r;
          const int s_ = wn + (j << 4) + nn;
          Gs[t_ * GSP + s_] = g[i][j][r];
        }
  }

  // ---- per-head: mask + Y = W * xh ----
  for (int h = 0; h < 8; ++h) {
    const float Ah = -expf(A_log[h]);
    const float Dh = Dp[h];
    __syncthreads();   // Gs ready (1st iter); prev head's reads done (others)
    if (tid < 64) dts[tid] = dtb[(base + tid) * 8 + h];
    for (int i = tid; i < 4096; i += 128) {
      const int l = i >> 6, s = i & 63;
      xT[s * SP + l] = (_Float16)xbc[(base + l) * 640 + h * 64 + s];
    }
    __syncthreads();
    if (tid == 0) {
      float c2 = 0.f;
      for (int l = 0; l < 64; ++l) { c2 += Ah * dts[l]; cum[l] = c2; }
    }
    __syncthreads();
    for (int i = tid; i < 4096; i += 128) {
      const int t = i >> 6, s = i & 63;
      float v = 0.f;
      if (s <= t) v = expf(cum[t] - cum[s]) * dts[s] * Gs[t * GSP + s];
      Ws[t * SP + s] = (_Float16)v;
    }
    __syncthreads();

    v8f y[2][2] = {{VZ, VZ}, {VZ, VZ}};
#pragma unroll
    for (int k0 = 0; k0 < 64; k0 += 32) {
      v16h a0 = frag_a(Ws, SP, wm, k0, lane);
      v16h a1 = frag_a(Ws, SP, wm + 16, k0, lane);
      v16h b0 = frag_b(xT, SP, wn, k0, lane);
      v16h b1 = frag_b(xT, SP, wn + 16, k0, lane);
      y[0][0] = wmma16(a0, b0, y[0][0]);
      y[0][1] = wmma16(a0, b1, y[0][1]);
      y[1][0] = wmma16(a1, b0, y[1][0]);
      y[1][1] = wmma16(a1, b1, y[1][1]);
    }
#pragma unroll
    for (int i = 0; i < 2; ++i)
#pragma unroll
      for (int j = 0; j < 2; ++j)
#pragma unroll
        for (int r = 0; r < 8; ++r) {
          const int t_ = wm + (i << 4) + (hh << 3) + r;
          const int p_ = wn + (j << 4) + nn;
          const float v = y[i][j][r] + Dh * (float)xT[p_ * SP + t_];
          yh[(base + t_) * 512 + h * 64 + p_] = v;
        }
  }
}

// ------------------------------ small kernels ------------------------------
__global__ void k_dft_init(float* __restrict__ tabs) {
  const int idx = blockIdx.x * blockDim.x + threadIdx.x;
  if (idx >= 4 * 4096) return;
  const int t = idx >> 12, a = (idx >> 6) & 63, b = idx & 63;
  const float ang = 6.283185307179586f * (float)(a * b) / 64.0f;
  const float inv8 = 0.125f;
  float v;
  if (t == 0) { const float sg = (b & 1) ? -1.f : 1.f; v =  sg * cosf(ang) * inv8; }
  else if (t == 1) { const float sg = (b & 1) ? -1.f : 1.f; v = -sg * sinf(ang) * inv8; }
  else if (t == 2) { const float sg = (a & 1) ? -1.f : 1.f; v =  sg * cosf(ang) * inv8; }
  else             { const float sg = (a & 1) ? -1.f : 1.f; v =  sg * sinf(ang) * inv8; }
  tabs[idx] = v;
}

__global__ void k_ln2d(const float* __restrict__ x, const float* __restrict__ w,
                       const float* __restrict__ b, float* __restrict__ o) {
  const int p = blockIdx.x * blockDim.x + threadIdx.x;
  if (p >= 32768) return;
  const int bb = p >> 12, hw = p & 4095;
  const size_t base = ((size_t)bb << 19) + hw;
  float s = 0.f;
  for (int c = 0; c < 128; ++c) s += x[base + ((size_t)c << 12)];
  const float mu = s * (1.0f / 128.0f);
  float vs = 0.f;
  for (int c = 0; c < 128; ++c) {
    const float d = x[base + ((size_t)c << 12)] - mu;
    vs += d * d;
  }
  const float rs = rsqrtf(vs * (1.0f / 128.0f) + EPS);
  for (int c = 0; c < 128; ++c)
    o[base + ((size_t)c << 12)] = (x[base + ((size_t)c << 12)] - mu) * rs * w[c] + b[c];
}

__global__ void k_tr64(const float* __restrict__ in, float* __restrict__ out) {
  const int idx = blockIdx.x * blockDim.x + threadIdx.x;
  if (idx >= 1024 * 4096) return;
  const int img = idx >> 12, p = (idx >> 6) & 63, q = idx & 63;
  out[((size_t)img << 12) + (p << 6) + q] = in[((size_t)img << 12) + (q << 6) + p];
}

__global__ void k_tr64tok(const float* __restrict__ in, float* __restrict__ tok, int coff) {
  const int idx = blockIdx.x * blockDim.x + threadIdx.x;
  if (idx >= 8 * 128 * 4096) return;
  const int c = idx & 127;
  int t = idx >> 7;
  const int p = t & 63; t >>= 6;
  const int q = t & 63; const int b = t >> 6;
  const float v = in[(((size_t)b * 128 + c) << 12) + (p << 6) + q];
  tok[((((size_t)b << 6) + q) * 64 + p) * 256 + coff + c] = v;
}

__global__ void k_tok2nchw(const float* __restrict__ tok, float* __restrict__ re,
                           float* __restrict__ im) {
  const int idx = blockIdx.x * blockDim.x + threadIdx.x;
  if (idx >= 8 * 128 * 4096) return;
  const int w = idx & 63, h = (idx >> 6) & 63, c = (idx >> 12) & 127, b = idx >> 19;
  const size_t tix = ((((size_t)b << 6) + h) * 64 + w) * 256;
  re[(((size_t)b * 128 + c) << 12) + (h << 6) + w] = tok[tix + c];
  im[(((size_t)b * 128 + c) << 12) + (h << 6) + w] = tok[tix + 128 + c];
}

__device__ __forceinline__ float band_mask(int h, int w, float lo, float hi, int last) {
  const float fy = (float)(h - 32) * (1.0f / 64.0f);
  const float fx = (float)(w - 32) * (1.0f / 64.0f);
  const float r = sqrtf(fy * fy + fx * fx) * 1.41421356237309515f;
  const bool in = (r >= lo) && (last ? (r <= hi) : (r < hi));
  return in ? 1.0f : 0.0f;
}

__global__ void k_maskmul(const float* __restrict__ in, float* __restrict__ out,
                          float lo, float hi, int last) {
  const int idx = blockIdx.x * blockDim.x + threadIdx.x;
  if (idx >= 32768 * 256) return;
  const int tok = idx >> 8;
  const int h = (tok >> 6) & 63, w = tok & 63;
  out[idx] = in[idx] * band_mask(h, w, lo, hi, last);
}

__global__ void k_bn_act(float* __restrict__ buf, int n, int Cm1,
                         const float* __restrict__ w, const float* __restrict__ b,
                         const float* __restrict__ rm, const float* __restrict__ rv,
                         int act) {
  const int idx = blockIdx.x * blockDim.x + threadIdx.x;
  if (idx >= n) return;
  const int c = idx & Cm1;
  const float s = w[c] * rsqrtf(rv[c] + EPS);
  float v = buf[idx] * s + (b[c] - rm[c] * s);
  if (act == 1) v = gelu_f(v);
  buf[idx] = v;
}

__global__ void k_trhw(const float* __restrict__ in, float* __restrict__ out) {
  const int idx = blockIdx.x * blockDim.x + threadIdx.x;
  if (idx >= 32768 * 256) return;
  const int c = idx & 255;
  int t = idx >> 8;
  const int w = t & 63, h = (t >> 6) & 63, b = t >> 12;
  out[((((size_t)b << 6) + w) * 64 + h) * 256 + c] = in[idx];
}

__global__ void k_ln_tok(const float* __restrict__ in, const float* __restrict__ w,
                         const float* __restrict__ b, float* __restrict__ out) {
  const int lane = threadIdx.x & 31;
  const int tok = blockIdx.x * 8 + (threadIdx.x >> 5);
  if (tok >= 32768) return;
  const size_t base = (size_t)tok * 256;
  float v[8];
#pragma unroll
  for (int k = 0; k < 8; ++k) v[k] = in[base + lane + 32 * k];
  float s = 0.f;
#pragma unroll
  for (int k = 0; k < 8; ++k) s += v[k];
  const float mu = wave_sum32(s) * (1.0f / 256.0f);
  float vs = 0.f;
#pragma unroll
  for (int k = 0; k < 8; ++k) { const float d = v[k] - mu; vs += d * d; }
  const float rs = rsqrtf(wave_sum32(vs) * (1.0f / 256.0f) + EPS);
#pragma unroll
  for (int k = 0; k < 8; ++k) {
    const int c = lane + 32 * k;
    out[base + c] = (v[k] - mu) * rs * w[c] + b[c];
  }
}

__global__ void k_dwconv_silu(const float* __restrict__ zx, const float* __restrict__ cw,
                              const float* __restrict__ cb, float* __restrict__ out) {
  const int idx = blockIdx.x * blockDim.x + threadIdx.x;
  if (idx >= 32768 * 640) return;
  const int c = idx % 640;
  const int tok = idx / 640;
  const int l = tok & 63;
  float acc = 0.f;
#pragma unroll
  for (int k = 0; k < 4; ++k) {
    const int back = 3 - k;
    if (l >= back) acc += cw[c * 4 + k] * zx[(size_t)(tok - back) * ZROW + 512 + c];
  }
  out[idx] = silu_f(acc + cb[c]);
}

__global__ void k_dt_prep(const float* __restrict__ zx, const float* __restrict__ dtb_p,
                          float* __restrict__ dt) {
  const int idx = blockIdx.x * blockDim.x + threadIdx.x;
  if (idx >= 32768 * 8) return;
  const int h = idx & 7, tok = idx >> 3;
  dt[idx] = softplus_f(zx[(size_t)tok * ZROW + 1152 + h] + dtb_p[h]);
}

__global__ void k_gate_rms(float* __restrict__ yh, const float* __restrict__ zx,
                           const float* __restrict__ nw) {
  const int lane = threadIdx.x & 31;
  const int tok = blockIdx.x * 8 + (threadIdx.x >> 5);
  if (tok >= 32768) return;
  const size_t yb = (size_t)tok * 512, zb = (size_t)tok * ZROW;
  float v[16];
  float ss = 0.f;
#pragma unroll
  for (int k = 0; k < 16; ++k) {
    const int c = lane + 32 * k;
    const float g = yh[yb + c] * silu_f(zx[zb + c]);
    v[k] = g; ss += g * g;
  }
  const float rs = rsqrtf(wave_sum32(ss) * (1.0f / 512.0f) + EPS);
#pragma unroll
  for (int k = 0; k < 16; ++k) {
    const int c = lane + 32 * k;
    yh[yb + c] = v[k] * rs * nw[c];
  }
}

__global__ void k_gate_combine(float* __restrict__ glin, const float* __restrict__ gb,
                               const float* __restrict__ yr, const float* __restrict__ yc) {
  const int idx = blockIdx.x * blockDim.x + threadIdx.x;
  if (idx >= 32768 * 256) return;
  const int c = idx & 255;
  const float g = sigm_f(glin[idx] + gb[c]);
  glin[idx] = g * yr[idx] + (1.0f - g) * yc[idx];
}

__global__ void k_bn_maskacc(const float* __restrict__ src, float* __restrict__ acc,
                             const float* __restrict__ w, const float* __restrict__ b,
                             const float* __restrict__ rm, const float* __restrict__ rv,
                             float lo, float hi, int last) {
  const int idx = blockIdx.x * blockDim.x + threadIdx.x;
  if (idx >= 32768 * 256) return;
  const int c = idx & 255;
  const int tok = idx >> 8;
  const int h = (tok >> 6) & 63, ww = tok & 63;
  const float s = w[c] * rsqrtf(rv[c] + EPS);
  const float v = src[idx] * s + (b[c] - rm[c] * s);
  acc[idx] += v * band_mask(h, ww, lo, hi, last);
}

__global__ void k_final(const float* __restrict__ x, const float* __restrict__ ot,
                        const float* __restrict__ w, const float* __restrict__ b,
                        const float* __restrict__ rm, const float* __restrict__ rv,
                        const float* __restrict__ gamma, float* __restrict__ out) {
  const int idx = blockIdx.x * blockDim.x + threadIdx.x;
  if (idx >= 8 * 128 * 4096) return;
  const int ww = idx & 63, h = (idx >> 6) & 63, c = (idx >> 12) & 127, bb = idx >> 19;
  const float o = ot[((((size_t)bb << 6) + h) * 64 + ww) * 128 + c];
  const float s = w[c] * rsqrtf(rv[c] + EPS);
  const float v = gelu_f(o * s + (b[c] - rm[c] * s));
  out[idx] = x[idx] + gamma[c] * v;
}

// ------------------------------- host driver -------------------------------
struct BandP {
  const float *pre_W, *pbw, *pbb, *pbrm, *pbrv;
  const float *post_W, *qbw, *qbb, *qbrm, *qbrv;
  const float *row_w, *row_b, *col_w, *col_b;
  const float *gate_W, *gate_b;
};
struct MambaP {
  const float *Win, *conv_w, *conv_b, *dt_bias, *A_log, *D, *norm_w, *Wout;
};

extern "C" void kernel_launch(void* const* d_in, const int* in_sizes, int n_in,
                              void* d_out, int out_size, void* d_ws, size_t ws_size,
                              hipStream_t stream) {
  (void)in_sizes; (void)out_size; (void)ws_size;
  if (n_in < 73) return;
  auto F = [&](int i) { return (const float*)d_in[i]; };

  // ---- parameter mapping (setup_inputs dict insertion order, flattened) ----
  int ii = 0;
  const float* x     = F(ii++);
  const float* pre_w = F(ii++);
  const float* pre_b = F(ii++);
  BandP bp[3];
  for (int i = 0; i < 3; ++i) {
    bp[i].pre_W = F(ii++);
    bp[i].pbw = F(ii++); bp[i].pbb = F(ii++); bp[i].pbrm = F(ii++); bp[i].pbrv = F(ii++);
    bp[i].post_W = F(ii++);
    bp[i].qbw = F(ii++); bp[i].qbb = F(ii++); bp[i].qbrm = F(ii++); bp[i].qbrv = F(ii++);
    bp[i].row_w = F(ii++); bp[i].row_b = F(ii++);
    bp[i].col_w = F(ii++); bp[i].col_b = F(ii++);
    bp[i].gate_W = F(ii++); bp[i].gate_b = F(ii++);
  }
  MambaP mp[2];
  for (int i = 0; i < 2; ++i) {
    mp[i].Win = F(ii++); mp[i].conv_w = F(ii++); mp[i].conv_b = F(ii++);
    mp[i].dt_bias = F(ii++); mp[i].A_log = F(ii++); mp[i].D = F(ii++);
    mp[i].norm_w = F(ii++); mp[i].Wout = F(ii++);
  }
  const float* sp_W  = F(ii++);
  const float* sbw = F(ii++); const float* sbb = F(ii++);
  const float* sbrm = F(ii++); const float* sbrv = F(ii++);
  const float* gamma = F(ii++);
  float* out = (float*)d_out;

  // ---- workspace bump allocator (floats) ----
  float* wsf = (float*)d_ws;
  size_t off = 0;
  auto alloc = [&](size_t n) { float* p = wsf + off; off += (n + 63) & ~(size_t)63; return p; };
  float* tabs  = alloc(4 * 4096);
  float* Fr = tabs, *Fi = tabs + 4096, *Gr = tabs + 8192, *Gi = tabs + 12288;
  float* WinP0 = alloc((size_t)ZROW * 256);   // zero-padded Win (row mamba)
  float* WinP1 = alloc((size_t)ZROW * 256);   // zero-padded Win (col mamba)
  float* xn    = alloc(4194304);              // ln2d out; reused as o_tok
  float* s1    = alloc(4194304);
  float* s2    = alloc(4194304);
  float* s3    = alloc(4194304);
  float* s4    = alloc(4194304);
  float* xp    = alloc(8388608);              // xp tokens; reused as sp tokens
  float* accT  = alloc(8388608);
  float* xb    = alloc(8388608);
  float* tdir  = alloc(8388608);
  float* tks   = alloc(8388608);
  float* yrT   = alloc(8388608);
  float* ycT   = alloc(8388608);
  float* t0    = alloc(8388608);
  float* zx    = alloc((size_t)32768 * ZROW);
  float* xbc   = alloc((size_t)32768 * 640);
  float* dtb   = alloc((size_t)32768 * 8);
  float* yh    = alloc((size_t)32768 * 512);

  auto gemm = [&](const float* A, int lda, const float* B, int ldb, float* C, int ldc,
                  int M, int N, int K, int flags) {
    dim3 g((unsigned)(N >> 6), (unsigned)(M >> 6));
    k_gemm<<<g, 128, 0, stream>>>(A, lda, B, ldb, C, ldc, K, flags);
  };
  #define EW(kern, n, ...) kern<<<((n) + 255) / 256, 256, 0, stream>>>(__VA_ARGS__)

  const float blo[3] = {0.0f, 0.25f, 0.5f};
  const float bhi[3] = {0.25f, 0.5f, 1.0f};

  // ---- padded in_proj weights (zero rows 1160..1215) ----
  hipMemsetAsync(WinP0, 0, (size_t)ZROW * 256 * sizeof(float), stream);
  hipMemsetAsync(WinP1, 0, (size_t)ZROW * 256 * sizeof(float), stream);
  hipMemcpyAsync(WinP0, mp[0].Win, (size_t)1160 * 256 * sizeof(float),
                 hipMemcpyDeviceToDevice, stream);
  hipMemcpyAsync(WinP1, mp[1].Win, (size_t)1160 * 256 * sizeof(float),
                 hipMemcpyDeviceToDevice, stream);
  const float* WinP[2] = {WinP0, WinP1};

  // ---- tables + pre-LN ----
  EW(k_dft_init, 4 * 4096, tabs);
  EW(k_ln2d, 32768, x, pre_w, pre_b, xn);

  // ---- forward FFT (DFT GEMMs, fftshift folded) ----
  gemm(xn, 64, Fr, 64, s1, 64, 65536, 64, 64, 0);
  gemm(xn, 64, Fi, 64, s2, 64, 65536, 64, 64, 0);
  EW(k_tr64, 1024 * 4096, s1, s3);
  EW(k_tr64, 1024 * 4096, s2, s4);
  gemm(s3, 64, Fr, 64, s1, 64, 65536, 64, 64, 0);
  gemm(s4, 64, Fi, 64, s1, 64, 65536, 64, 64, 3);
  gemm(s3, 64, Fi, 64, s2, 64, 65536, 64, 64, 0);
  gemm(s4, 64, Fr, 64, s2, 64, 65536, 64, 64, 1);
  EW(k_tr64tok, 8 * 128 * 4096, s1, xp, 0);
  EW(k_tr64tok, 8 * 128 * 4096, s2, xp, 128);

  hipMemsetAsync(accT, 0, (size_t)8388608 * sizeof(float), stream);

  // ---- per-band processing (token layout [b,h,w,256]) ----
  for (int bi = 0; bi < 3; ++bi) {
    const BandP& P = bp[bi];
    EW(k_maskmul, 32768 * 256, xp, t0, blo[bi], bhi[bi], bi == 2);
    gemm(t0, 256, P.pre_W, 256, xb, 256, 32768, 256, 256, 0);
    EW(k_bn_act, 32768 * 256, xb, 32768 * 256, 255, P.pbw, P.pbb, P.pbrm, P.pbrv, 1);

    for (int d = 0; d < 2; ++d) {
      const MambaP& Mq = mp[d];
      const float* src = xb;
      if (d == 1) { EW(k_trhw, 32768 * 256, xb, tdir); src = tdir; }
      k_ln_tok<<<4096, 256, 0, stream>>>(src, d ? P.col_w : P.row_w,
                                         d ? P.col_b : P.row_b, tks);
      gemm(tks, 256, WinP[d], 256, zx, ZROW, 32768, ZROW, 256, 0);
      EW(k_dwconv_silu, 32768 * 640, zx, Mq.conv_w, Mq.conv_b, xbc);
      EW(k_dt_prep, 32768 * 8, zx, Mq.dt_bias, dtb);
      k_ssd<<<512, 128, 0, stream>>>(xbc, dtb, Mq.A_log, Mq.D, yh);
      k_gate_rms<<<4096, 256, 0, stream>>>(yh, zx, Mq.norm_w);
      if (d == 0) {
        gemm(yh, 512, Mq.Wout, 512, yrT, 256, 32768, 256, 512, 0);
      } else {
        gemm(yh, 512, Mq.Wout, 512, tdir, 256, 32768, 256, 512, 0);
        EW(k_trhw, 32768 * 256, tdir, ycT);
      }
    }
    gemm(yrT, 256, P.gate_W, 512, t0, 256, 32768, 256, 256, 0);
    gemm(ycT, 256, P.gate_W + 256, 512, t0, 256, 32768, 256, 256, 1);
    EW(k_gate_combine, 32768 * 256, t0, P.gate_b, yrT, ycT);
    gemm(t0, 256, P.post_W, 256, tks, 256, 32768, 256, 256, 0);
    EW(k_bn_maskacc, 32768 * 256, tks, accT, P.qbw, P.qbb, P.qbrm, P.qbrv,
       blo[bi], bhi[bi], bi == 2);
  }

  // ---- inverse FFT ----
  EW(k_tok2nchw, 8 * 128 * 4096, accT, s1, s2);
  gemm(s1, 64, Gr, 64, s3, 64, 65536, 64, 64, 0);
  gemm(s2, 64, Gi, 64, s3, 64, 65536, 64, 64, 3);
  gemm(s1, 64, Gi, 64, s4, 64, 65536, 64, 64, 0);
  gemm(s2, 64, Gr, 64, s4, 64, 65536, 64, 64, 1);
  EW(k_tr64, 1024 * 4096, s3, s1);
  EW(k_tr64, 1024 * 4096, s4, s2);
  gemm(s1, 64, Gr, 64, s3, 64, 65536, 64, 64, 0);
  gemm(s2, 64, Gi, 64, s3, 64, 65536, 64, 64, 3);
  gemm(s1, 64, Gi, 64, s4, 64, 65536, 64, 64, 0);
  gemm(s2, 64, Gr, 64, s4, 64, 65536, 64, 64, 1);
  EW(k_tr64tok, 8 * 128 * 4096, s3, xp, 0);
  EW(k_tr64tok, 8 * 128 * 4096, s4, xp, 128);

  // ---- spatial projection + residual ----
  gemm(xp, 256, sp_W, 256, xn, 128, 32768, 128, 256, 0);
  EW(k_final, 8 * 128 * 4096, x, xn, sbw, sbb, sbrm, sbrv, gamma, out);
  #undef EW
}
